// GCNPredictor_56135222559006
// MI455X (gfx1250) — matrix operand order repaired and checked by reading, not compiled, not run on hardware
//
#include <hip/hip_runtime.h>
#include <hip/hip_bf16.h>

typedef __bf16 bf16;
typedef __attribute__((ext_vector_type(16))) __bf16 v16bf;
typedef __attribute__((ext_vector_type(8)))  __bf16 v8bf;
typedef __attribute__((ext_vector_type(8)))  float  v8f;
typedef __attribute__((ext_vector_type(4)))  unsigned int v4u;
typedef __attribute__((ext_vector_type(8)))  int v8i_t;
typedef __attribute__((ext_vector_type(4)))  int v4i_t;

#define HD 128
#define BN_EPS 1e-5f

// ---------------------------------------------------------------- utilities

__global__ void k_cvt_bf16(const float* __restrict__ in, bf16* __restrict__ out, int n) {
  int i = blockIdx.x * blockDim.x + threadIdx.x;
  if (i < n) out[i] = (bf16)in[i];
}

__global__ void k_init_deg(float* __restrict__ deg, int n) {
  int i = blockIdx.x * blockDim.x + threadIdx.x;
  if (i < n) deg[i] = 1.0f;  // self-loop weight contributes 1 to every degree
}

__global__ void k_edge_deg(const int* __restrict__ col, const float* __restrict__ w,
                           float* __restrict__ deg, int E) {
  int e = blockIdx.x * blockDim.x + threadIdx.x;
  if (e < E) atomicAdd(&deg[col[e]], w[e]);
}

__global__ void k_rsqrt_deg(float* __restrict__ deg, int n) {
  int i = blockIdx.x * blockDim.x + threadIdx.x;
  if (i < n) { float d = deg[i]; deg[i] = (d > 0.f) ? rsqrtf(d) : 0.f; }
}

// Pack a row-major 128x128 f32 weight into the WMMA B-operand (32x16 K-major
// tiles, bf16).  Element i of lane (16*half+n) of tile (kc,nt) is
// W[kc*32 + (i<8 ? 8*half + i : 16 + 8*half + (i-8))][nt*16 + n]   (CDNA5 ISA 7.12.2 / 7.12.4).
__global__ void k_pack_w(const float* __restrict__ W, bf16* __restrict__ P) {
  int idx = blockIdx.x * blockDim.x + threadIdx.x;   // 4*8*32*16 = 16384
  int i    = idx & 15;
  int lane = (idx >> 4) & 31;
  int t    = idx >> 9;            // kc*8 + nt
  int kc   = t >> 3;
  int nt   = t & 7;
  int half = lane >> 4;
  int n    = lane & 15;
  int k    = kc * 32 + ((i < 8) ? (8 * half + i) : (16 + 8 * half + (i - 8)));
  P[idx] = (bf16)W[k * HD + nt * 16 + n];
}

// ------------------------------------------------------------- WMMA GEMM
// C[N x 128] = A[N x 128](bf16) * B[128 x 128](pre-packed bf16), f32 accum.
// Packed B (32 KB) is staged into LDS once per block via the Tensor Data
// Mover (one tensor_load_to_lds issued by wave 0, TENSORcnt-waited), then
// each wave computes a 16-row tile: 8 col tiles x 4 K-chunks = 32 WMMA ops,
// with B operands fed from ds_load_b128.
__global__ void __launch_bounds__(256) k_gemm_wmma(const bf16* __restrict__ A,
                                                   const bf16* __restrict__ Bp,
                                                   float* __restrict__ C, int ntiles) {
  __shared__ __align__(128) bf16 Blds[4 * 8 * 32 * 16];   // 32 KB packed B

  // --- TDM stage: wave 0 DMAs the packed weight into LDS ---
  if ((threadIdx.x >> 5) == 0) {
    unsigned long long ga = (unsigned long long)Bp;
    unsigned lds_base = (unsigned)(size_t)(&Blds[0]);     // LDS byte address (addr[31:0])
    const unsigned elems = 4096;                          // 32768 B / 8 B units
    v4u g0;
    g0[0] = 1u;                                           // count=1, user descriptor
    g0[1] = lds_base;                                     // D#.lds_addr
    g0[2] = (unsigned)(ga & 0xffffffffull);               // global_addr[31:0]
    g0[3] = (unsigned)((ga >> 32) & 0x01ffffffull)        // global_addr[56:32]
            | (2u << 30);                                 // type = 2 ("image")
    v8i_t g1;
    g1[0] = (int)(3u << 16);                              // workgroup_mask=0, data_size=8B
    g1[1] = (int)(elems << 16);                           // tensor_dim0[15:0]
    g1[2] = (int)((elems >> 16) | (1u << 16));            // tensor_dim0[31:16] | tensor_dim1=1
    g1[3] = (int)(elems << 16);                           // tile_dim0 = 4096
    g1[4] = 1;                                            // tile_dim1 = 1
    g1[5] = (int)elems;                                   // tensor_dim0_stride[31:0]
    g1[6] = 0;                                            // stride hi / dim1_stride lo
    g1[7] = 0;
    v4i_t z4 = {0, 0, 0, 0};                              // groups 2/3 unused (<=2D)
    v8i_t z8 = {0, 0, 0, 0, 0, 0, 0, 0};
    __builtin_amdgcn_tensor_load_to_lds(g0, g1, z4, z4, z8, 0);
    __builtin_amdgcn_s_wait_tensorcnt((short)0);
  }
  __syncthreads();

  int gwave = (int)((blockIdx.x * blockDim.x + threadIdx.x) >> 5);
  if (gwave >= ntiles) return;              // wave-uniform: EXEC stays all-1s
  int lane = threadIdx.x & 31;
  int half = lane >> 4;
  int mr   = lane & 15;

  const bf16* __restrict__ arow = A + (size_t)(gwave * 16 + mr) * HD;
  const v16bf* bl = (const v16bf*)Blds;

  v8f acc[8];
#pragma unroll
  for (int t = 0; t < 8; ++t)
#pragma unroll
    for (int r = 0; r < 8; ++r) acc[t][r] = 0.f;

#pragma unroll
  for (int kc = 0; kc < 4; ++kc) {
    // A operand: row mr, K = kc*32 + {8*half..8*half+7, 16+8*half..16+8*half+7}
    v8bf alo = *(const v8bf*)(arow + kc * 32 + 8 * half);
    v8bf ahi = *(const v8bf*)(arow + kc * 32 + 16 + 8 * half);
    v16bf a;
#pragma unroll
    for (int i = 0; i < 8; ++i) { a[i] = alo[i]; a[8 + i] = ahi[i]; }
#pragma unroll
    for (int nt = 0; nt < 8; ++nt) {
      v16bf b = bl[(kc * 8 + nt) * 32 + lane];            // ds_load from staged B
      acc[nt] = __builtin_amdgcn_wmma_f32_16x16x32_bf16(
          false, a, false, b, (short)0, acc[nt], false, false);
    }
  }
  // D layout: VGPR r -> row (gwave*16 + 8*half + r), col (nt*16 + mr)
#pragma unroll
  for (int nt = 0; nt < 8; ++nt) {
    float* __restrict__ crow = C + (size_t)(gwave * 16 + half * 8) * HD + nt * 16 + mr;
#pragma unroll
    for (int r = 0; r < 8; ++r) crow[(size_t)r * HD] = acc[nt][r];
  }
}

// ----------------------------------------------------- message passing
// agg[i] = hw[i]*dinv[i]^2 + bias  (self-loop, norm = dinv*1*dinv; also
// re-initializes the accumulation buffer deterministically each call)
__global__ void k_selfloop_bias(const float* __restrict__ hw, const float* __restrict__ dinv,
                                const float* __restrict__ bias, float* __restrict__ agg, int n) {
  int idx = blockIdx.x * blockDim.x + threadIdx.x;
  if (idx < n) {
    int i = idx >> 7, f = idx & 127;
    float d = dinv[i];
    agg[idx] = hw[idx] * d * d + bias[f];
  }
}

// One wave per edge, float4 per lane (32*4 = 128 features), atomic scatter-add.
__global__ void k_edge_scatter(const int* __restrict__ row, const int* __restrict__ col,
                               const float* __restrict__ w, const float* __restrict__ dinv,
                               const float* __restrict__ hw, float* __restrict__ agg, int E) {
  int gid = blockIdx.x * blockDim.x + threadIdx.x;
  int e = gid >> 5;
  if (e >= E) return;
  int lane = gid & 31;
  int r = row[e], c = col[e];
  float nrm = dinv[r] * w[e] * dinv[c];
  const float4 v = ((const float4*)(hw + (size_t)r * HD))[lane];
  float* dst = agg + (size_t)c * HD + lane * 4;
  atomicAdd(dst + 0, v.x * nrm);
  atomicAdd(dst + 1, v.y * nrm);
  atomicAdd(dst + 2, v.z * nrm);
  atomicAdd(dst + 3, v.w * nrm);
}

// -------------------------------------------------------------- batchnorm
__global__ void k_zero_stats(float* __restrict__ s, float* __restrict__ sq) {
  int i = threadIdx.x;
  if (i < HD) { s[i] = 0.f; sq[i] = 0.f; }
}

__global__ void k_col_stats(const float* __restrict__ h, float* __restrict__ s,
                            float* __restrict__ sq, int N) {
  int f = threadIdx.x;            // 128 threads, column f
  int b = blockIdx.x;             // 256 blocks stride over rows
  float acc = 0.f, acc2 = 0.f;
  for (int r = b; r < N; r += 256) {
    float v = h[(size_t)r * HD + f];
    acc += v; acc2 += v * v;
  }
  atomicAdd(&s[f], acc);
  atomicAdd(&sq[f], acc2);
}

// normalize + affine + relu; write f32 (for head) and bf16 (next GEMM A)
__global__ void k_bn_relu(const float* __restrict__ h, const float* __restrict__ s,
                          const float* __restrict__ sq, const float* __restrict__ g,
                          const float* __restrict__ be, float* __restrict__ outf,
                          bf16* __restrict__ outb, int n, float invN) {
  int idx = blockIdx.x * blockDim.x + threadIdx.x;
  if (idx < n) {
    int f = idx & 127;
    float mu  = s[f] * invN;
    float var = sq[f] * invN - mu * mu;          // biased, matches reference
    float v = (h[idx] - mu) * rsqrtf(var + BN_EPS) * g[f] + be[f];
    v = fmaxf(v, 0.f);
    outf[idx] = v;
    outb[idx] = (bf16)v;
  }
}

// -------------------------------------------------------------- head
// out[0] = relu(h[0]@Wh+bh) @ Wa + ba ;  out[1] = relu(h[1]@Wh+bh) @ Wz + bz
__global__ void k_head(const float* __restrict__ h, const float* __restrict__ Wh,
                       const float* __restrict__ bh, const float* __restrict__ Wa,
                       const float* __restrict__ ba, const float* __restrict__ Wz,
                       const float* __restrict__ bz, float* __restrict__ out) {
  __shared__ float s0[HD], s1[HD];
  int j = threadIdx.x;  // 0..127
  float a0 = 0.f, a1 = 0.f;
  for (int k = 0; k < HD; ++k) {
    float wk = Wh[k * HD + j];
    a0 += h[k] * wk;
    a1 += h[HD + k] * wk;
  }
  a0 = fmaxf(a0 + bh[j], 0.f);
  a1 = fmaxf(a1 + bh[j], 0.f);
  s0[j] = a0 * Wa[j];
  s1[j] = a1 * Wz[j];
  __syncthreads();
  for (int stride = 64; stride > 0; stride >>= 1) {
    if (j < stride) { s0[j] += s0[j + stride]; s1[j] += s1[j + stride]; }
    __syncthreads();
  }
  if (j == 0) { out[0] = s0[0] + ba[0]; out[1] = s1[0] + bz[0]; }
}

// -------------------------------------------------------------- launch

extern "C" void kernel_launch(void* const* d_in, const int* in_sizes, int n_in,
                              void* d_out, int out_size, void* d_ws, size_t ws_size,
                              hipStream_t stream) {
  const float* x  = (const float*)d_in[0];
  const int*   ei = (const int*)d_in[1];
  const float* ew = (const float*)d_in[2];
  const float* W[3]  = {(const float*)d_in[3], (const float*)d_in[5], (const float*)d_in[7]};
  const float* bb[3] = {(const float*)d_in[4], (const float*)d_in[6], (const float*)d_in[8]};
  const float* g[3]  = {(const float*)d_in[9],  (const float*)d_in[11], (const float*)d_in[13]};
  const float* be[3] = {(const float*)d_in[10], (const float*)d_in[12], (const float*)d_in[14]};
  const float* Wh = (const float*)d_in[15];
  const float* bh = (const float*)d_in[16];
  const float* Wa = (const float*)d_in[17];
  const float* ba = (const float*)d_in[18];
  const float* Wz = (const float*)d_in[19];
  const float* bz = (const float*)d_in[20];

  const int N = in_sizes[0] / HD;
  const int E = in_sizes[2];
  const int* rowi = ei;
  const int* coli = ei + E;

  // workspace carve-up (256B aligned)
  char* ws = (char*)d_ws;
  size_t off = 0;
  auto carve = [&](size_t bytes) {
    off = (off + 255) & ~(size_t)255;
    void* p = ws + off;
    off += bytes;
    return p;
  };
  float* buf0   = (float*)carve((size_t)N * HD * sizeof(float));  // GEMM out / normalized h
  float* buf1   = (float*)carve((size_t)N * HD * sizeof(float));  // scatter accumulator
  bf16*  Xbf    = (bf16*) carve((size_t)N * HD * sizeof(bf16));   // GEMM A operand
  float* dinv   = (float*)carve((size_t)N * sizeof(float));
  bf16*  pack   = (bf16*) carve((size_t)4 * 8 * 32 * 16 * sizeof(bf16)); // packed weight
  float* colsum = (float*)carve(HD * sizeof(float));
  float* colsq  = (float*)carve(HD * sizeof(float));
  (void)ws_size; (void)n_in; (void)out_size;

  const int nfeat  = N * HD;
  const int ntiles = N / 16;  // N = 50000 -> 3125 full tiles
  dim3 b256(256);

  // degrees + symmetric normalization
  k_cvt_bf16 <<<(nfeat + 255) / 256, b256, 0, stream>>>(x, Xbf, nfeat);
  k_init_deg <<<(N + 255) / 256,     b256, 0, stream>>>(dinv, N);
  k_edge_deg <<<(E + 255) / 256,     b256, 0, stream>>>(coli, ew, dinv, E);
  k_rsqrt_deg<<<(N + 255) / 256,     b256, 0, stream>>>(dinv, N);

  for (int L = 0; L < 3; ++L) {
    k_pack_w       <<<64, b256, 0, stream>>>(W[L], pack);
    k_gemm_wmma    <<<(ntiles * 32 + 255) / 256, b256, 0, stream>>>(Xbf, pack, buf0, ntiles);
    k_selfloop_bias<<<(nfeat + 255) / 256, b256, 0, stream>>>(buf0, dinv, bb[L], buf1, nfeat);
    k_edge_scatter <<<((size_t)E * 32 + 255) / 256, b256, 0, stream>>>(rowi, coli, ew, dinv,
                                                                       buf0, buf1, E);
    k_zero_stats   <<<1, b256, 0, stream>>>(colsum, colsq);
    k_col_stats    <<<256, dim3(HD), 0, stream>>>(buf1, colsum, colsq, N);
    k_bn_relu      <<<(nfeat + 255) / 256, b256, 0, stream>>>(buf1, colsum, colsq, g[L], be[L],
                                                              buf0, Xbf, nfeat, 1.0f / (float)N);
  }

  k_head<<<1, dim3(HD), 0, stream>>>(buf0, Wh, bh, Wa, ba, Wz, bz, (float*)d_out);
}
